// PolicyHead_7318624272517
// MI455X (gfx1250) — matrix operand order repaired and checked by reading, not compiled, not run on hardware
//
#include <hip/hip_runtime.h>
#include <hip/hip_bf16.h>
#include <math.h>

// ---------------------------------------------------------------------------
// Model dims (fixed by the reference)
// ---------------------------------------------------------------------------
#define LAYERS 4
#define DMODEL 512
#define NHEAD  8
#define HDIM   64
#define FFDIM  1024
#define BATCH  64
#define SEQ    64
#define ROWS   (BATCH * SEQ)      // 4096 token rows
#define LN_EPS 1e-5f

typedef unsigned short u16;
typedef unsigned int   u32;

typedef __bf16 bf16_t;
typedef bf16_t v16bf __attribute__((ext_vector_type(16)));
typedef float  v8f   __attribute__((ext_vector_type(8)));

// 32-byte WMMA operand fragment, loadable as two 16B global loads
union Frag32 {
    uint4 q[2];
    v16bf v;
};

__device__ __forceinline__ u16 f32_to_bf16(float f) {
    u32 u = __float_as_uint(f);
    u32 r = u + 0x7FFFu + ((u >> 16) & 1u);   // round-to-nearest-even
    return (u16)(r >> 16);
}

__device__ __forceinline__ float gelu_exact(float x) {
    return 0.5f * x * (1.0f + erff(x * 0.70710678118654752f));
}

// ---------------------------------------------------------------------------
// Elementwise f32 -> bf16 conversion (weights + initial embeddings only)
// ---------------------------------------------------------------------------
__global__ __launch_bounds__(256)
void k_f32_to_bf16(const float* __restrict__ in, u16* __restrict__ out, int n) {
    int i = blockIdx.x * 256 + threadIdx.x;
    if (i < n) out[i] = f32_to_bf16(in[i]);
}

// ---------------------------------------------------------------------------
// WMMA GEMM:  C[M,N] = act( A[M,K](bf16) * W[N,K]^T(bf16) + bias[N] )
//
// Register-blocked: each wave owns a 32x64 strip of C (2 M-tiles x 4 N-tiles
// = 8 f32 accumulators).  Per K-step (32) it loads 2 A frags + 4 B frags and
// issues 8 independent v_wmma_f32_16x16x32_bf16 -> ~21 FLOP/byte from L2 and
// deep latency hiding.  Block = 4 waves in a 2x2 arrangement covering a
// 64x128 C tile.  grid = (N/128, M/64).
//
// Epilogue: +bias, optional exact GELU, then optional f32 and/or bf16 store
// (bf16 store feeds the next GEMM directly, no separate conversion pass).
// Fragment layout per CDNA5 ISA 7.12.2 (bf16 A 16x32, B 32x16, C f32 16x16).
// ---------------------------------------------------------------------------
__global__ __launch_bounds__(128)
void k_gemm_bf16_wmma(const u16* __restrict__ A, int lda,
                      const u16* __restrict__ W, int ldw,
                      const float* __restrict__ bias,
                      float* __restrict__ C, u16* __restrict__ Cbf,
                      int ldc, int K, int act)
{
    const int lane = threadIdx.x & 31;
    const int wave = threadIdx.x >> 5;
    const int mSub = wave >> 1;            // 0..1
    const int nSub = wave & 1;             // 0..1

    const int m0 = blockIdx.y * 64 + mSub * 32;   // wave's first M row
    const int n0 = blockIdx.x * 128 + nSub * 64;  // wave's first N col

    const int half = lane >> 4;     // 0: lanes 0-15, 1: lanes 16-31
    const int l16  = lane & 15;

    // A: lane holds row (mtile+l16), K chunks [half*8, +8) and [+16, +8)
    const u16* aPtr0 = A + (size_t)(m0 + l16) * lda + half * 8;
    const u16* aPtr1 = aPtr0 + (size_t)16 * lda;
    // B: lane holds W row (= output col ntile+l16), K chunks [half*8, +8) and [+16, +8)
    const u16* bPtrBase = W + (size_t)(n0 + l16) * ldw + half * 8;
    const size_t bStep = (size_t)16 * ldw;

    v8f acc[2][4];
#pragma unroll
    for (int mi = 0; mi < 2; ++mi)
#pragma unroll
        for (int ni = 0; ni < 4; ++ni)
            acc[mi][ni] = (v8f){0.f, 0.f, 0.f, 0.f, 0.f, 0.f, 0.f, 0.f};

    for (int k0 = 0; k0 < K; k0 += 32) {
        Frag32 fa[2], fb[4];
        fa[0].q[0] = *(const uint4*)(aPtr0 + k0);
        fa[0].q[1] = *(const uint4*)(aPtr0 + k0 + 16);
        fa[1].q[0] = *(const uint4*)(aPtr1 + k0);
        fa[1].q[1] = *(const uint4*)(aPtr1 + k0 + 16);
#pragma unroll
        for (int ni = 0; ni < 4; ++ni) {
            const u16* bp = bPtrBase + ni * bStep + k0;
            fb[ni].q[0] = *(const uint4*)(bp);
            fb[ni].q[1] = *(const uint4*)(bp + 16);
        }
#pragma unroll
        for (int mi = 0; mi < 2; ++mi)
#pragma unroll
            for (int ni = 0; ni < 4; ++ni)
                acc[mi][ni] = __builtin_amdgcn_wmma_f32_16x16x32_bf16(
                    false, fa[mi].v, false, fb[ni].v,
                    (short)0, acc[mi][ni], false, false);
    }

    // C layout: VGPR r -> row mtile + half*8 + r, col ntile + l16
#pragma unroll
    for (int ni = 0; ni < 4; ++ni) {
        const int col = n0 + ni * 16 + l16;
        const float bv = bias ? bias[col] : 0.0f;
#pragma unroll
        for (int mi = 0; mi < 2; ++mi) {
            const int rowBase = m0 + mi * 16 + half * 8;
#pragma unroll
            for (int r = 0; r < 8; ++r) {
                float v = acc[mi][ni][r] + bv;
                if (act == 1) v = gelu_exact(v);
                const size_t idx = (size_t)(rowBase + r) * ldc + col;
                if (C)   C[idx]   = v;
                if (Cbf) Cbf[idx] = f32_to_bf16(v);
            }
        }
    }
}

// ---------------------------------------------------------------------------
// Attention: one block (64 threads) per (b, h). qkv rows are 3*D wide with
// q at +0, k at +512, v at +1024 (head h occupies cols h*64..h*64+63).
// Thread t owns query row t: scores/softmax/AV fully in registers,
// K & V tiles staged in LDS (2 * 64*65*4 = 33 KB).  Writes the attended
// output both as f32 (residual path) and bf16 (next-GEMM operand).
// ---------------------------------------------------------------------------
__global__ __launch_bounds__(64)
void k_attention(const float* __restrict__ qkv,
                 float* __restrict__ o, u16* __restrict__ obf)
{
    __shared__ float Ks[SEQ][HDIM + 1];
    __shared__ float Vs[SEQ][HDIM + 1];

    const int b = blockIdx.x >> 3;
    const int h = blockIdx.x & 7;
    const int t = threadIdx.x;             // 0..63

    const float* base = qkv + (size_t)(b * SEQ) * (3 * DMODEL) + h * HDIM;
    const float* rowT = base + (size_t)t * (3 * DMODEL);

#pragma unroll 4
    for (int d = 0; d < HDIM; ++d) {
        Ks[t][d] = rowT[DMODEL + d];
        Vs[t][d] = rowT[2 * DMODEL + d];
    }
    float q[HDIM];
#pragma unroll 4
    for (int d = 0; d < HDIM; ++d) q[d] = rowT[d];
    __syncthreads();

    float p[SEQ];
    float mx = -3.0e38f;
    for (int j = 0; j < SEQ; ++j) {
        float s = 0.f;
#pragma unroll 4
        for (int d = 0; d < HDIM; ++d) s += q[d] * Ks[j][d];
        s *= 0.125f;                        // 1/sqrt(HDIM)
        p[j] = s;
        mx = fmaxf(mx, s);
    }
    float sum = 0.f;
    for (int j = 0; j < SEQ; ++j) { p[j] = __expf(p[j] - mx); sum += p[j]; }
    const float inv = 1.0f / sum;

    float out[HDIM];
#pragma unroll 4
    for (int d = 0; d < HDIM; ++d) out[d] = 0.f;
    for (int j = 0; j < SEQ; ++j) {
        const float pj = p[j] * inv;
#pragma unroll 4
        for (int d = 0; d < HDIM; ++d) out[d] += pj * Vs[j][d];
    }

    const size_t rowOff = (size_t)(b * SEQ + t) * DMODEL + h * HDIM;
#pragma unroll 4
    for (int d = 0; d < HDIM; ++d) {
        o[rowOff + d]   = out[d];
        obf[rowOff + d] = f32_to_bf16(out[d]);
    }
}

// ---------------------------------------------------------------------------
// x = LayerNorm(x + r) * g + b, emitted as f32 (residual) AND bf16 (next
// GEMM operand).  One wave32 per 512-wide row (__shfl_xor reduction);
// block = (32,8) -> 8 rows/block, grid = ROWS/8.
// ---------------------------------------------------------------------------
__global__ __launch_bounds__(256)
void k_res_layernorm(float* __restrict__ x, const float* __restrict__ r,
                     const float* __restrict__ g, const float* __restrict__ bta,
                     u16* __restrict__ xb)
{
    const int row  = blockIdx.x * 8 + threadIdx.y;
    const int lane = threadIdx.x;          // 0..31

    float* xr = x + (size_t)row * DMODEL;
    const float* rr = r + (size_t)row * DMODEL;
    u16* xbr = xb + (size_t)row * DMODEL;

    float v[16];
    float s = 0.f, ss = 0.f;
#pragma unroll
    for (int i = 0; i < 16; ++i) {
        const int c = lane + i * 32;
        const float t = xr[c] + rr[c];
        v[i] = t; s += t; ss += t * t;
    }
#pragma unroll
    for (int m = 16; m >= 1; m >>= 1) {
        s  += __shfl_xor(s,  m, 32);
        ss += __shfl_xor(ss, m, 32);
    }
    const float mean = s * (1.0f / DMODEL);
    const float var  = ss * (1.0f / DMODEL) - mean * mean;
    const float rstd = rsqrtf(var + LN_EPS);
#pragma unroll
    for (int i = 0; i < 16; ++i) {
        const int c = lane + i * 32;
        const float y = (v[i] - mean) * rstd * g[c] + bta[c];
        xr[c]  = y;
        xbr[c] = f32_to_bf16(y);
    }
}

// ---------------------------------------------------------------------------
// Policy head: logits[b,f,t] = sum_h gelu(f[b,f,h] + t[b,t,h] + bm1[h]) *
//                              wm2[h] + bm2[0]
// block = 256 threads handles (b, 4 f-rows, all 64 t). t tile (64x64) staged
// in LDS with +1 pad -> conflict-free tt[tx][hh] reads. grid = (B, S/4).
// ---------------------------------------------------------------------------
__global__ __launch_bounds__(256)
void k_policy_combine(const float* __restrict__ f, const float* __restrict__ t,
                      const float* __restrict__ bm1, const float* __restrict__ wm2,
                      const float* __restrict__ bm2, float* __restrict__ out)
{
    __shared__ float tt[64][65];
    __shared__ float fb[4][64];
    __shared__ float wS[64];

    const int b   = blockIdx.x;
    const int fg  = blockIdx.y;            // f group of 4 rows
    const int tid = threadIdx.x;
    const int ty  = tid >> 6;              // 0..3  -> f sub-row
    const int tx  = tid & 63;              // 0..63 -> t index
    const int f_idx = fg * 4 + ty;

    const float* fRow  = f + (size_t)(b * SEQ + f_idx) * FFDIM;
    const float* tBase = t + (size_t)(b * SEQ) * FFDIM;

    float acc = 0.f;
    for (int h0 = 0; h0 < FFDIM; h0 += 64) {
        // cooperative load: 64x64 t tile, coalesced along h
#pragma unroll
        for (int rr = 0; rr < 16; ++rr) {
            const int r = ty * 16 + rr;
            tt[r][tx] = tBase[(size_t)r * FFDIM + h0 + tx];
        }
        fb[ty][tx] = fRow[h0 + tx] + bm1[h0 + tx];
        if (tid < 64) wS[tid] = wm2[h0 + tid];
        __syncthreads();

#pragma unroll 4
        for (int hh = 0; hh < 64; ++hh) {
            const float xg = fb[ty][hh] + tt[tx][hh];
            acc += gelu_exact(xg) * wS[hh];
        }
        __syncthreads();
    }
    out[(size_t)(b * SEQ + f_idx) * SEQ + tx] = acc + bm2[0];
}

// ---------------------------------------------------------------------------
// Host orchestration
// ---------------------------------------------------------------------------
static void convert_to_bf16(const float* in, u16* out, int n, hipStream_t s) {
    k_f32_to_bf16<<<(n + 255) / 256, 256, 0, s>>>(in, out, n);
}

static void gemm(const u16* A, int lda, const u16* W, int ldw,
                 const float* bias, float* C, u16* Cbf, int ldc,
                 int M, int N, int K, int act, hipStream_t s) {
    dim3 grid(N / 128, M / 64);
    k_gemm_bf16_wmma<<<grid, 128, 0, s>>>(A, lda, W, ldw, bias, C, Cbf,
                                          ldc, K, act);
}

extern "C" void kernel_launch(void* const* d_in, const int* in_sizes, int n_in,
                              void* d_out, int out_size, void* d_ws, size_t ws_size,
                              hipStream_t stream) {
    (void)in_sizes; (void)n_in; (void)out_size; (void)ws_size;

    const float* emb   = (const float*)d_in[0];   // (B,S,D)
    const float* Wqkv  = (const float*)d_in[1];   // (L,3D,D)
    const float* bqkv  = (const float*)d_in[2];   // (L,3D)
    const float* Wo    = (const float*)d_in[3];   // (L,D,D)
    const float* bo    = (const float*)d_in[4];   // (L,D)
    const float* ln1_g = (const float*)d_in[5];
    const float* ln1_b = (const float*)d_in[6];
    const float* ln2_g = (const float*)d_in[7];
    const float* ln2_b = (const float*)d_in[8];
    const float* W1    = (const float*)d_in[9];   // (L,FF,D)
    const float* b1    = (const float*)d_in[10];  // (L,FF)
    const float* W2    = (const float*)d_in[11];  // (L,D,FF)
    const float* b2    = (const float*)d_in[12];  // (L,D)
    const float* Wm1   = (const float*)d_in[13];  // (FF,2D)
    const float* bm1   = (const float*)d_in[14];  // (FF,)
    const float* Wm2   = (const float*)d_in[15];  // (1,FF)
    const float* bm2   = (const float*)d_in[16];  // (1,)
    float* out = (float*)d_out;

    // -------- workspace bump allocator (256B aligned) --------
    char* ws = (char*)d_ws;
    size_t off = 0;
    auto alloc = [&](size_t bytes) -> char* {
        char* p = ws + off;
        off = (off + bytes + 255) & ~(size_t)255;
        return p;
    };

    float* x     = (float*)alloc((size_t)ROWS * DMODEL * 4);        //  8 MB
    float* proj  = (float*)alloc((size_t)ROWS * DMODEL * 4);        //  8 MB
    float* qkvb  = (float*)alloc((size_t)ROWS * 3 * DMODEL * 4);    // 24 MB
    float* tbuf  = (float*)alloc((size_t)ROWS * FFDIM * 4);         // 16 MB
    u16*   xb    = (u16*)  alloc((size_t)ROWS * DMODEL * 2);        //  4 MB
    u16*   hb    = (u16*)  alloc((size_t)ROWS * FFDIM * 2);         //  8 MB

    u16* Wqkv_bf = (u16*)alloc((size_t)LAYERS * 3 * DMODEL * DMODEL * 2);
    u16* Wo_bf   = (u16*)alloc((size_t)LAYERS * DMODEL * DMODEL * 2);
    u16* W1_bf   = (u16*)alloc((size_t)LAYERS * FFDIM * DMODEL * 2);
    u16* W2_bf   = (u16*)alloc((size_t)LAYERS * DMODEL * FFDIM * 2);
    u16* Wm1_bf  = (u16*)alloc((size_t)FFDIM * 2 * DMODEL * 2);

    // policy-head f buffer reuses the dead qkv buffer (24 MB >= 16 MB)
    float* fbuf = qkvb;

    // -------- one-time (per launch) weight conversion to bf16 --------
    convert_to_bf16(Wqkv, Wqkv_bf, LAYERS * 3 * DMODEL * DMODEL, stream);
    convert_to_bf16(Wo,   Wo_bf,   LAYERS * DMODEL * DMODEL,     stream);
    convert_to_bf16(W1,   W1_bf,   LAYERS * FFDIM * DMODEL,      stream);
    convert_to_bf16(W2,   W2_bf,   LAYERS * DMODEL * FFDIM,      stream);
    convert_to_bf16(Wm1,  Wm1_bf,  FFDIM * 2 * DMODEL,           stream);

    // x = node_embeddings (f32 residual stream) + bf16 operand copy
    hipMemcpyAsync(x, emb, (size_t)ROWS * DMODEL * 4,
                   hipMemcpyDeviceToDevice, stream);
    convert_to_bf16(emb, xb, ROWS * DMODEL, stream);

    // -------- transformer layers --------
    for (int l = 0; l < LAYERS; ++l) {
        // qkv = x @ Wqkv[l]^T + bqkv[l]
        gemm(xb, DMODEL,
             Wqkv_bf + (size_t)l * 3 * DMODEL * DMODEL, DMODEL,
             bqkv + (size_t)l * 3 * DMODEL,
             qkvb, nullptr, 3 * DMODEL, ROWS, 3 * DMODEL, DMODEL, 0, stream);

        // multi-head softmax attention -> proj (f32) + xb (bf16 operand)
        k_attention<<<BATCH * NHEAD, 64, 0, stream>>>(qkvb, proj, xb);

        // attended = o @ Wo[l]^T + bo[l]
        gemm(xb, DMODEL,
             Wo_bf + (size_t)l * DMODEL * DMODEL, DMODEL,
             bo + (size_t)l * DMODEL,
             proj, nullptr, DMODEL, ROWS, DMODEL, DMODEL, 0, stream);

        // x = LN(x + attended); also emits xb (bf16)
        k_res_layernorm<<<ROWS / 8, dim3(32, 8), 0, stream>>>(
            x, proj, ln1_g + (size_t)l * DMODEL, ln1_b + (size_t)l * DMODEL, xb);

        // h = gelu(x @ W1[l]^T + b1[l])  -> bf16 only (only FF2 consumes it)
        gemm(xb, DMODEL,
             W1_bf + (size_t)l * FFDIM * DMODEL, DMODEL,
             b1 + (size_t)l * FFDIM,
             nullptr, hb, FFDIM, ROWS, FFDIM, DMODEL, 1, stream);

        // ff = h @ W2[l]^T + b2[l]
        gemm(hb, FFDIM,
             W2_bf + (size_t)l * DMODEL * FFDIM, FFDIM,
             b2 + (size_t)l * DMODEL,
             proj, nullptr, DMODEL, ROWS, DMODEL, FFDIM, 0, stream);

        // x = LN(x + ff); also emits xb (bf16)
        k_res_layernorm<<<ROWS / 8, dim3(32, 8), 0, stream>>>(
            x, proj, ln2_g + (size_t)l * DMODEL, ln2_b + (size_t)l * DMODEL, xb);
    }

    // -------- policy head --------
    // f = x @ Wm1[:, :D]^T ; t = x @ Wm1[:, D:]^T   (no bias; bm1 fused later)
    gemm(xb, DMODEL, Wm1_bf,          2 * DMODEL, nullptr,
         fbuf, nullptr, FFDIM, ROWS, FFDIM, DMODEL, 0, stream);
    gemm(xb, DMODEL, Wm1_bf + DMODEL, 2 * DMODEL, nullptr,
         tbuf, nullptr, FFDIM, ROWS, FFDIM, DMODEL, 0, stream);

    // logits[b,f,t] = sum_h gelu(f+t+bm1)*Wm2 + bm2
    k_policy_combine<<<dim3(BATCH, SEQ / 4), 256, 0, stream>>>(
        fbuf, tbuf, bm1, Wm2, bm2, out);
}
